// GeoGCNConv_31894427140229
// MI455X (gfx1250) — matrix operand order repaired and checked by reading, not compiled, not run on hardware
//
#include <hip/hip_runtime.h>
#include <hip/hip_bf16.h>

typedef __attribute__((ext_vector_type(16))) __bf16 v16bf;
typedef __attribute__((ext_vector_type(8)))  __bf16 v8bf;
typedef __attribute__((ext_vector_type(8)))  float  v8f;

// ---------------------------------------------------------------------------
// Graph preprocessing kernels
// ---------------------------------------------------------------------------
__global__ void k_fill1(float* __restrict__ p, int n) {
  int i = blockIdx.x * blockDim.x + threadIdx.x;
  if (i < n) p[i] = 1.0f;               // self-loop weight contributes +1 to deg
}

__global__ void k_deg_accum(float* __restrict__ deg, const int* __restrict__ dst,
                            const float* __restrict__ w, int e) {
  int i = blockIdx.x * blockDim.x + threadIdx.x;
  if (i < e) atomicAdd(&deg[dst[i]], w[i]);
}

__global__ void k_rsqrt_inplace(float* __restrict__ p, int n) {
  int i = blockIdx.x * blockDim.x + threadIdx.x;
  if (i < n) p[i] = rsqrtf(p[i]);       // deg >= 1 always (self loop)
}

__global__ void k_edge_norm(float* __restrict__ norm, const float* __restrict__ dinv,
                            const int* __restrict__ src, const int* __restrict__ dst,
                            const float* __restrict__ w, int e) {
  int i = blockIdx.x * blockDim.x + threadIdx.x;
  if (i < e) norm[i] = dinv[src[i]] * w[i] * dinv[dst[i]];
}

// agg[i,c] = dinv[i]^2 * h[i,c] + bias[c]   (self-loop term + bias, also zero-init)
__global__ void k_self_bias(float* __restrict__ agg, const float* __restrict__ h,
                            const float* __restrict__ dinv, const float* __restrict__ bias,
                            long total, int f) {
  long idx = (long)blockIdx.x * blockDim.x + threadIdx.x;
  if (idx >= total) return;
  int i = (int)(idx / f);
  int c = (int)(idx % f);
  float di = dinv[i];
  agg[idx] = di * di * h[idx] + bias[c];
}

// one thread per (edge, feature): gather h1[src] row, scale, atomic-scatter to dst
__global__ void k_scatter_edges(float* __restrict__ agg, const float* __restrict__ h,
                                const float* __restrict__ norm, const int* __restrict__ src,
                                const int* __restrict__ dst, long total, int shift,
                                int stride, int fvalid) {
  long idx = (long)blockIdx.x * blockDim.x + threadIdx.x;
  if (idx >= total) return;
  int e = (int)(idx >> shift);
  int c = (int)(idx & ((1 << shift) - 1));
  if (c >= fvalid) return;
  float v = norm[e] * h[(long)src[e] * stride + c];
  atomicAdd(&agg[(long)dst[e] * stride + c], v);
}

__global__ void k_relu(float* __restrict__ p, long n) {
  long i = (long)blockIdx.x * blockDim.x + threadIdx.x;
  if (i < n) p[i] = fmaxf(p[i], 0.0f);
}

__global__ void k_logsoftmax10(const float* __restrict__ h, float* __restrict__ out, int n) {
  int i = blockIdx.x * blockDim.x + threadIdx.x;
  if (i >= n) return;
  float v[10];
  float m = -3.402823466e38f;
#pragma unroll
  for (int c = 0; c < 10; ++c) { v[c] = h[i * 10 + c]; m = fmaxf(m, v[c]); }
  float s = 0.0f;
#pragma unroll
  for (int c = 0; c < 10; ++c) s += expf(v[c] - m);
  float l = logf(s);
#pragma unroll
  for (int c = 0; c < 10; ++c) out[i * 10 + c] = v[c] - m - l;
}

// ---------------------------------------------------------------------------
// GEMM1: h1[n,64] = feat[n,128] @ W1[128,64]   (bf16 WMMA, f32 accumulate)
// Block = 256 threads = 8 waves; each wave owns one 16-row M tile, all 4 N tiles.
// W1 staged once per block into LDS, pre-swizzled into B-fragment lane order:
//   fragment (kk,t): element [lane*16 + j] = W1[kk*32 + (lane>=16?8:0) + (j&7) + (j>=8?16:0)][t*16 + (lane&15)]
// ---------------------------------------------------------------------------
__global__ __launch_bounds__(256) void gemm1_wmma(const float* __restrict__ feat,
                                                  const float* __restrict__ W1,
                                                  float* __restrict__ h1, int n) {
  __shared__ __bf16 sW[128 * 64];
  for (int idx = threadIdx.x; idx < 128 * 64; idx += 256) {
    int kkt = idx >> 9;            // fragment id: kk*4 + t
    int rem = idx & 511;
    int lane = rem >> 4;
    int j = rem & 15;
    int kk = kkt >> 2, t = kkt & 3;
    int k = kk * 32 + ((lane >> 4) << 3) + (j & 7) + ((j >> 3) << 4);
    int col = t * 16 + (lane & 15);
    sW[idx] = (__bf16)W1[k * 64 + col];
  }
  __syncthreads();

  const int wave = threadIdx.x >> 5;
  const int lane = threadIdx.x & 31;
  const int m0 = (blockIdx.x * 8 + wave) * 16;
  if (m0 >= n) return;                       // uniform per wave

  const int lhalf = lane >> 4;               // 0: lanes 0-15, 1: lanes 16-31
  const int l15 = lane & 15;

  v8f acc0 = {}, acc1 = {}, acc2 = {}, acc3 = {};

  int row = m0 + l15;
  if (row >= n) row = n - 1;                 // clamp tail so EXEC stays all-ones
  const float* frow = feat + (long)row * 128;

#pragma unroll
  for (int kk = 0; kk < 4; ++kk) {
    const int kb = kk * 32 + lhalf * 8;      // A layout: lanes<16 K=0..7,16..23; lanes>=16 +8
    const float4 f0 = *(const float4*)(frow + kb);
    const float4 f1 = *(const float4*)(frow + kb + 4);
    const float4 f2 = *(const float4*)(frow + kb + 16);
    const float4 f3 = *(const float4*)(frow + kb + 20);
    v16bf a;
    a[0] = (__bf16)f0.x;  a[1] = (__bf16)f0.y;  a[2] = (__bf16)f0.z;  a[3] = (__bf16)f0.w;
    a[4] = (__bf16)f1.x;  a[5] = (__bf16)f1.y;  a[6] = (__bf16)f1.z;  a[7] = (__bf16)f1.w;
    a[8] = (__bf16)f2.x;  a[9] = (__bf16)f2.y;  a[10] = (__bf16)f2.z; a[11] = (__bf16)f2.w;
    a[12] = (__bf16)f3.x; a[13] = (__bf16)f3.y; a[14] = (__bf16)f3.z; a[15] = (__bf16)f3.w;

#pragma unroll
    for (int t = 0; t < 4; ++t) {
      const __bf16* bp = &sW[(((kk << 2) | t) << 9) + (lane << 4)];
      v8bf blo = *(const v8bf*)bp;
      v8bf bhi = *(const v8bf*)(bp + 8);
      v16bf b = __builtin_shufflevector(blo, bhi, 0, 1, 2, 3, 4, 5, 6, 7,
                                        8, 9, 10, 11, 12, 13, 14, 15);
      v8f c = (t == 0) ? acc0 : (t == 1) ? acc1 : (t == 2) ? acc2 : acc3;
      c = __builtin_amdgcn_wmma_f32_16x16x32_bf16(false, a, false, b, (short)0, c,
                                                  false, false);
      if (t == 0) acc0 = c; else if (t == 1) acc1 = c; else if (t == 2) acc2 = c; else acc3 = c;
    }
  }

  // C/D layout: lane (l15) = column, VGPR r = row r (lanes<16) / r+8 (lanes>=16)
#pragma unroll
  for (int t = 0; t < 4; ++t) {
    const v8f c = (t == 0) ? acc0 : (t == 1) ? acc1 : (t == 2) ? acc2 : acc3;
    const int col = t * 16 + l15;
#pragma unroll
    for (int r = 0; r < 8; ++r) {
      int orow = m0 + lhalf * 8 + r;
      if (orow < n) h1[(long)orow * 64 + col] = c[r];
    }
  }
}

// ---------------------------------------------------------------------------
// GEMM2: h2[n,10] = x[n,64] @ W2[64,10]  (single zero-padded 16-wide N tile)
// ---------------------------------------------------------------------------
__global__ __launch_bounds__(256) void gemm2_wmma(const float* __restrict__ x,
                                                  const float* __restrict__ W2,
                                                  float* __restrict__ h2, int n) {
  __shared__ __bf16 sW[64 * 16];
  for (int idx = threadIdx.x; idx < 64 * 16; idx += 256) {
    int kk = idx >> 9;
    int rem = idx & 511;
    int lane = rem >> 4;
    int j = rem & 15;
    int k = kk * 32 + ((lane >> 4) << 3) + (j & 7) + ((j >> 3) << 4);
    int col = lane & 15;
    sW[idx] = (__bf16)((col < 10) ? W2[k * 10 + col] : 0.0f);
  }
  __syncthreads();

  const int wave = threadIdx.x >> 5;
  const int lane = threadIdx.x & 31;
  const int m0 = (blockIdx.x * 8 + wave) * 16;
  if (m0 >= n) return;

  const int lhalf = lane >> 4;
  const int l15 = lane & 15;

  v8f acc = {};
  int row = m0 + l15;
  if (row >= n) row = n - 1;
  const float* xr = x + (long)row * 64;

#pragma unroll
  for (int kk = 0; kk < 2; ++kk) {
    const int kb = kk * 32 + lhalf * 8;
    const float4 f0 = *(const float4*)(xr + kb);
    const float4 f1 = *(const float4*)(xr + kb + 4);
    const float4 f2 = *(const float4*)(xr + kb + 16);
    const float4 f3 = *(const float4*)(xr + kb + 20);
    v16bf a;
    a[0] = (__bf16)f0.x;  a[1] = (__bf16)f0.y;  a[2] = (__bf16)f0.z;  a[3] = (__bf16)f0.w;
    a[4] = (__bf16)f1.x;  a[5] = (__bf16)f1.y;  a[6] = (__bf16)f1.z;  a[7] = (__bf16)f1.w;
    a[8] = (__bf16)f2.x;  a[9] = (__bf16)f2.y;  a[10] = (__bf16)f2.z; a[11] = (__bf16)f2.w;
    a[12] = (__bf16)f3.x; a[13] = (__bf16)f3.y; a[14] = (__bf16)f3.z; a[15] = (__bf16)f3.w;

    const __bf16* bp = &sW[(kk << 9) + (lane << 4)];
    v8bf blo = *(const v8bf*)bp;
    v8bf bhi = *(const v8bf*)(bp + 8);
    v16bf b = __builtin_shufflevector(blo, bhi, 0, 1, 2, 3, 4, 5, 6, 7,
                                      8, 9, 10, 11, 12, 13, 14, 15);
    acc = __builtin_amdgcn_wmma_f32_16x16x32_bf16(false, a, false, b, (short)0, acc,
                                                  false, false);
  }

#pragma unroll
  for (int r = 0; r < 8; ++r) {
    int orow = m0 + lhalf * 8 + r;
    if (orow < n && l15 < 10) h2[(long)orow * 10 + l15] = acc[r];
  }
}

// ---------------------------------------------------------------------------
// Launch: deg -> dinv -> norm -> GEMM1 -> self+bias -> scatter -> relu ->
//         GEMM2 -> self+bias -> scatter -> log_softmax
// ---------------------------------------------------------------------------
extern "C" void kernel_launch(void* const* d_in, const int* in_sizes, int n_in,
                              void* d_out, int out_size, void* d_ws, size_t ws_size,
                              hipStream_t stream) {
  const float* feat = (const float*)d_in[0];   // [n,128]
  const int*   eidx = (const int*)d_in[1];     // [2,E]
  const float* ew   = (const float*)d_in[2];   // [E]
  const float* W1   = (const float*)d_in[3];   // [128,64]
  const float* b1   = (const float*)d_in[4];   // [64]
  const float* W2   = (const float*)d_in[5];   // [64,10]
  const float* b2   = (const float*)d_in[6];   // [10]
  float* out = (float*)d_out;                  // [n,10]

  const int n = in_sizes[0] / 128;
  const int E = in_sizes[2];
  const int* src = eidx;
  const int* dst = eidx + E;

  // workspace layout (floats)
  float* ws   = (float*)d_ws;
  float* dinv = ws;                               // n        (deg, then rsqrt in place)
  float* norm = dinv + n;                         // E
  float* h1   = norm + E;                         // n*64
  float* agg1 = h1 + (size_t)n * 64;              // n*64     (becomes x after relu)
  float* h2   = agg1 + (size_t)n * 64;            // n*10
  float* agg2 = h2 + (size_t)n * 10;              // n*10

  auto cdiv = [](long a, long b) -> unsigned { return (unsigned)((a + b - 1) / b); };

  // degree / norm
  k_fill1<<<cdiv(n, 256), 256, 0, stream>>>(dinv, n);
  k_deg_accum<<<cdiv(E, 256), 256, 0, stream>>>(dinv, dst, ew, E);
  k_rsqrt_inplace<<<cdiv(n, 256), 256, 0, stream>>>(dinv, n);
  k_edge_norm<<<cdiv(E, 256), 256, 0, stream>>>(norm, dinv, src, dst, ew, E);

  // layer 1
  gemm1_wmma<<<cdiv(n, 128), 256, 0, stream>>>(feat, W1, h1, n);
  k_self_bias<<<cdiv((long)n * 64, 256), 256, 0, stream>>>(agg1, h1, dinv, b1,
                                                           (long)n * 64, 64);
  k_scatter_edges<<<cdiv((long)E * 64, 256), 256, 0, stream>>>(agg1, h1, norm, src, dst,
                                                               (long)E * 64, 6, 64, 64);
  k_relu<<<cdiv((long)n * 64, 256), 256, 0, stream>>>(agg1, (long)n * 64);

  // layer 2
  gemm2_wmma<<<cdiv(n, 128), 256, 0, stream>>>(agg1, W2, h2, n);
  k_self_bias<<<cdiv((long)n * 10, 256), 256, 0, stream>>>(agg2, h2, dinv, b2,
                                                           (long)n * 10, 10);
  k_scatter_edges<<<cdiv((long)E * 16, 256), 256, 0, stream>>>(agg2, h2, norm, src, dst,
                                                               (long)E * 16, 4, 10, 10);

  // output
  k_logsoftmax10<<<cdiv(n, 256), 256, 0, stream>>>(agg2, out, n);
}
